// AttentionLayer_71116068487952
// MI455X (gfx1250) — matrix operand order repaired and checked by reading, not compiled, not run on hardware
//
#include <hip/hip_runtime.h>

// ---------------------------------------------------------------------------
// MI455X (gfx1250) attention-with-sink-competition.
// All GEMMs run on v_wmma_f32_16x16x32_bf16 (wave32). sim/attn are never
// materialized: column-softmax stats + EPS renormalization are folded
// algebraically into a flash-style two-pass over the key axis.
// ---------------------------------------------------------------------------

typedef __bf16 bf16;
typedef __attribute__((ext_vector_type(8)))  __bf16 v8bf;
typedef __attribute__((ext_vector_type(16))) __bf16 v16bf;
typedef __attribute__((ext_vector_type(8)))  float  v8f;

#define EPSV 1e-6f

__device__ __forceinline__ int lane_id() { return threadIdx.x & 31; }

__device__ __forceinline__ v16bf make_v16(v8bf lo, v8bf hi) {
  v16bf r;
#pragma unroll
  for (int i = 0; i < 8; ++i) { r[i] = lo[i]; r[i + 8] = hi[i]; }
  return r;
}

// A fragment (16x32 bf16) from row-major A with leading dim ld.
// Lane l: row = row0 + (l&15); VGPR0-3 hold K[kb..kb+7], VGPR4-7 K[kb+16..kb+23],
// kb = (l>>4)*8  -> two contiguous 16B loads.
__device__ __forceinline__ v16bf fragA(const bf16* A, int ld, int row0, int k0) {
  int l = lane_id();
  const bf16* p = A + (size_t)(row0 + (l & 15)) * ld + k0 + ((l >> 4) << 3);
  v8bf lo = *(const v8bf*)p;
  v8bf hi = *(const v8bf*)(p + 16);
  return make_v16(lo, hi);
}

// B fragment (32x16 bf16) loaded from B^T (N x K row-major, leading dim ld).
// Lane l: col = col0 + (l&15); holds K[kb..kb+7], K[kb+16..kb+23], kb = (l>>4)*8.
__device__ __forceinline__ v16bf fragB(const bf16* BT, int ld, int col0, int k0) {
  int l = lane_id();
  const bf16* p = BT + (size_t)(col0 + (l & 15)) * ld + k0 + ((l >> 4) << 3);
  v8bf lo = *(const v8bf*)p;
  v8bf hi = *(const v8bf*)(p + 16);
  return make_v16(lo, hi);
}

__device__ __forceinline__ v8f wmma_bf16(v16bf a, v16bf b, v8f c) {
  return __builtin_amdgcn_wmma_f32_16x16x32_bf16(false, a, false, b, (short)0, c,
                                                 false, false);
}

// ---------------------------------------------------------------------------
// Weight transpose + fp32->bf16 convert: W [512][N] -> WT [N][512]
// ---------------------------------------------------------------------------
__global__ void transpose_to_bf16_kernel(const float* __restrict__ W,
                                         bf16* __restrict__ WT, int N) {
  int idx = blockIdx.x * blockDim.x + threadIdx.x;
  int n = idx >> 9, kk = idx & 511;
  WT[idx] = (bf16)W[(size_t)kk * N + n];
}

// ---------------------------------------------------------------------------
// Row LayerNorm (dim=512) -> bf16. One wave per row; lane owns 16 contiguous.
// ---------------------------------------------------------------------------
__global__ void ln_to_bf16_kernel(const float* __restrict__ x,
                                  const float* __restrict__ gamma,
                                  const float* __restrict__ beta,
                                  bf16* __restrict__ y, int nrows) {
  int w = blockIdx.x * (blockDim.x >> 5) + (threadIdx.x >> 5);
  if (w >= nrows) return;
  int l = lane_id();
  const float4* xp = (const float4*)(x + (size_t)w * 512 + l * 16);
  float v[16];
#pragma unroll
  for (int i = 0; i < 4; ++i) {
    float4 t = xp[i];
    v[i * 4 + 0] = t.x; v[i * 4 + 1] = t.y; v[i * 4 + 2] = t.z; v[i * 4 + 3] = t.w;
  }
  float s = 0.f;
#pragma unroll
  for (int i = 0; i < 16; ++i) s += v[i];
#pragma unroll
  for (int m = 1; m < 32; m <<= 1) s += __shfl_xor(s, m);
  float mu = s * (1.0f / 512.0f);
  float var = 0.f;
#pragma unroll
  for (int i = 0; i < 16; ++i) { float d = v[i] - mu; var += d * d; }
#pragma unroll
  for (int m = 1; m < 32; m <<= 1) var += __shfl_xor(var, m);
  float inv = rsqrtf(var * (1.0f / 512.0f) + EPSV);
  bf16 o[16];
#pragma unroll
  for (int i = 0; i < 16; ++i) {
    int c = l * 16 + i;
    o[i] = (bf16)((v[i] - mu) * inv * gamma[c] + beta[c]);
  }
  v8bf* yp = (v8bf*)(y + (size_t)w * 512 + l * 16);
  yp[0] = *(v8bf*)&o[0];
  yp[1] = *(v8bf*)&o[8];
}

// ---------------------------------------------------------------------------
// Generic GEMM: C[M,N] = A[M,512]bf16 @ (BT[N,512]bf16)^T + bias.
// 256 threads = 8 waves; wave -> 32x64 tile (2 A-frags x 4 B-frags = 8 WMMAs
// per k-step, B traffic amortized over 2 WMMAs); block -> 256x64 tile.
// ---------------------------------------------------------------------------
template <typename OUT>
__global__ void gemm_bf16_wmma_kernel(const bf16* __restrict__ A,
                                      const bf16* __restrict__ BT,
                                      const float* __restrict__ bias,
                                      OUT* __restrict__ C, int N, int K) {
  int wave = threadIdx.x >> 5;
  int l = lane_id();
  int m0 = blockIdx.x * 256 + wave * 32;
  int n0 = blockIdx.y * 64;
  v8f acc[2][4] = {};
  for (int kt = 0; kt < K; kt += 32) {
    // stream the next k-slice toward the caches (global_prefetch_b8)
    __builtin_prefetch(A + (size_t)(m0 + (l & 15)) * K + kt + 64, 0, 1);
    v16bf a0 = fragA(A, K, m0, kt);
    v16bf a1 = fragA(A, K, m0 + 16, kt);
#pragma unroll
    for (int nt = 0; nt < 4; ++nt) {
      v16bf b = fragB(BT, K, n0 + nt * 16, kt);
      acc[0][nt] = wmma_bf16(a0, b, acc[0][nt]);
      acc[1][nt] = wmma_bf16(a1, b, acc[1][nt]);
    }
  }
#pragma unroll
  for (int mt = 0; mt < 2; ++mt) {
    int rb = m0 + mt * 16 + ((l >> 4) << 3);
#pragma unroll
    for (int nt = 0; nt < 4; ++nt) {
      int n = n0 + nt * 16 + (l & 15);
      float bb = bias[n];
#pragma unroll
      for (int r = 0; r < 8; ++r)
        C[(size_t)(rb + r) * N + n] = (OUT)(acc[mt][nt][r] + bb);
    }
  }
}

// ---------------------------------------------------------------------------
// KV projection: CLN[8192,512] @ WkvT^T + bkv. Epilogue splits:
//   cols 0..511   -> K  row-major [b][4096][512]
//   cols 512..1023-> V^T [b][8][64][4096] (D-layout already holds 8 consecutive
//                    rows per lane -> contiguous along j: packed b128 stores)
// ---------------------------------------------------------------------------
__global__ void kv_gemm_kernel(const bf16* __restrict__ A,
                               const bf16* __restrict__ BT,
                               const float* __restrict__ bias,
                               bf16* __restrict__ Kout,
                               bf16* __restrict__ VT) {
  int wave = threadIdx.x >> 5;
  int l = lane_id();
  int m0 = blockIdx.x * 256 + wave * 32;
  int n0 = blockIdx.y * 64;
  v8f acc[2][4] = {};
  for (int kt = 0; kt < 512; kt += 32) {
    __builtin_prefetch(A + (size_t)(m0 + (l & 15)) * 512 + kt + 64, 0, 1);
    v16bf a0 = fragA(A, 512, m0, kt);
    v16bf a1 = fragA(A, 512, m0 + 16, kt);
#pragma unroll
    for (int nt = 0; nt < 4; ++nt) {
      v16bf b = fragB(BT, 512, n0 + nt * 16, kt);
      acc[0][nt] = wmma_bf16(a0, b, acc[0][nt]);
      acc[1][nt] = wmma_bf16(a1, b, acc[1][nt]);
    }
  }
#pragma unroll
  for (int mt = 0; mt < 2; ++mt) {
    int rb = m0 + mt * 16 + ((l >> 4) << 3);
#pragma unroll
    for (int nt = 0; nt < 4; ++nt) {
      int n = n0 + nt * 16 + (l & 15);
      float bb = bias[n];
      if (n < 512) {
#pragma unroll
        for (int r = 0; r < 8; ++r)
          Kout[(size_t)(rb + r) * 512 + n] = (bf16)(acc[mt][nt][r] + bb);
      } else {
        // rows rb..rb+7 are consecutive j -> pack one 16B store into V^T
        int c = n - 512;                 // = h*64 + dd
        int b = rb >> 12, j = rb & 4095;
        bf16 o[8];
#pragma unroll
        for (int r = 0; r < 8; ++r) o[r] = (bf16)(acc[mt][nt][r] + bb);
        *(v8bf*)(VT + ((size_t)(b * 512 + c)) * 4096 + j) = *(v8bf*)o;
      }
    }
  }
}

// ---------------------------------------------------------------------------
// Pass 1: per-(b,h,key j) column max / sum-exp over all 1024 queries.
// One wave per 16-key strip; online rescaled accumulation over 64 i-tiles.
// Column reduction is 8 in-lane adds + one shfl_xor(16) (WMMA D-layout).
// ---------------------------------------------------------------------------
__global__ void col_stats_kernel(const bf16* __restrict__ Q,
                                 const bf16* __restrict__ Kmat,
                                 float* __restrict__ colmax,
                                 float* __restrict__ colsum) {
  int wid = blockIdx.x * (blockDim.x >> 5) + (threadIdx.x >> 5);
  int bh = wid >> 8, jt = wid & 255;
  int b = bh >> 3, h = bh & 7;
  int l = lane_id();
  const bf16* qb = Q + (size_t)b * 1024 * 512 + h * 64;
  const bf16* kb = Kmat + (size_t)b * 4096 * 512 + h * 64;
  int j0 = jt * 16;
  float m = -3.0e38f, ssum = 0.f;
  for (int it = 0; it < 64; ++it) {
    __builtin_prefetch(qb + (size_t)(it * 16 + 16 + (l & 15)) * 512, 0, 1);
    v8f acc = {};
    acc = wmma_bf16(fragA(qb, 512, it * 16, 0),  fragB(kb, 512, j0, 0),  acc);
    acc = wmma_bf16(fragA(qb, 512, it * 16, 32), fragB(kb, 512, j0, 32), acc);
    float sv[8], tm = -3.0e38f;
#pragma unroll
    for (int r = 0; r < 8; ++r) { sv[r] = acc[r] * 0.125f; tm = fmaxf(tm, sv[r]); }
    tm = fmaxf(tm, __shfl_xor(tm, 16));          // combine lane-halves (same col)
    float nm = fmaxf(m, tm);
    float ls = 0.f;
#pragma unroll
    for (int r = 0; r < 8; ++r) ls += __expf(sv[r] - nm);
    ssum = ssum * __expf(m - nm) + ls;
    m = nm;
  }
  ssum += __shfl_xor(ssum, 16);
  if (l < 16) {
    colmax[bh * 4096 + j0 + l] = m;
    colsum[bh * 4096 + j0 + l] = ssum;
  }
}

// ---------------------------------------------------------------------------
// vsum[bh][dd] = sum_j V (needed for the +EPS term). One wave per V^T row.
// ---------------------------------------------------------------------------
__global__ void vsum_kernel(const bf16* __restrict__ VT, float* __restrict__ vsum) {
  int wid = blockIdx.x * (blockDim.x >> 5) + (threadIdx.x >> 5);
  if (wid >= 1024) return;
  int l = lane_id();
  const v8bf* p = (const v8bf*)(VT + (size_t)wid * 4096) + l;
  float s = 0.f;
#pragma unroll 4
  for (int it = 0; it < 16; ++it) {
    v8bf v = p[it * 32];
#pragma unroll
    for (int e = 0; e < 8; ++e) s += (float)v[e];
  }
#pragma unroll
  for (int m = 1; m < 32; m <<= 1) s += __shfl_xor(s, m);
  if (l == 0) vsum[wid] = s;
}

// ---------------------------------------------------------------------------
// Pass 2: out = (P@V + EPS*vsum) / (rowsum(P) + 4096*EPS).
// One wave per 16-query tile; loops 128 chunks of 32 keys. Score tiles go
// D-layout -> LDS -> A-layout for the PV WMMAs against V^T.
// Each wave owns a private LDS region and LDS ops are HW-in-order per wave
// (DScnt), so the D->A transpose needs only a compiler code-motion fence
// (wave_barrier, zero instructions) instead of block-wide s_barrier pairs.
// ---------------------------------------------------------------------------
__global__ void attn_out_kernel(const bf16* __restrict__ Q,
                                const bf16* __restrict__ Kmat,
                                const bf16* __restrict__ VT,
                                const float* __restrict__ colmax,
                                const float* __restrict__ colsum,
                                const float* __restrict__ vsum,
                                bf16* __restrict__ AO) {
  __shared__ __align__(16) bf16 lds_p[4][16][32];
  int wib = threadIdx.x >> 5;
  int wid = blockIdx.x * 4 + wib;
  int bh = wid >> 6, it = wid & 63;
  int b = bh >> 3, h = bh & 7;
  int l = lane_id();
  const bf16* qb  = Q    + (size_t)b * 1024 * 512 + h * 64;
  const bf16* kb  = Kmat + (size_t)b * 4096 * 512 + h * 64;
  const bf16* vtb = VT   + (size_t)bh * 64 * 4096;
  const float* cmax = colmax + (size_t)bh * 4096;
  const float* csum = colsum + (size_t)bh * 4096;

  v16bf aq0 = fragA(qb, 512, it * 16, 0);     // Q tile is loop-invariant
  v16bf aq1 = fragA(qb, 512, it * 16, 32);

  v8f oacc[4] = {};
  float rs[8] = {};
  int row_lo = (l >> 4) << 3;

  for (int jc = 0; jc < 128; ++jc) {
    // prefetch next chunk of K rows and V^T columns
    __builtin_prefetch(kb + (size_t)(jc * 32 + 32 + (l & 15)) * 512, 0, 1);
    __builtin_prefetch(vtb + (size_t)(l & 15) * 4096 + jc * 32 + 32, 0, 1);
#pragma unroll
    for (int half = 0; half < 2; ++half) {
      int j0 = jc * 32 + half * 16;
      v8f acc = {};
      acc = wmma_bf16(aq0, fragB(kb, 512, j0, 0),  acc);
      acc = wmma_bf16(aq1, fragB(kb, 512, j0, 32), acc);
      int j = j0 + (l & 15);
      float cm = cmax[j];
      float ics = 1.0f / csum[j];
#pragma unroll
      for (int r = 0; r < 8; ++r) {
        float p = __expf(acc[r] * 0.125f - cm) * ics;
        rs[r] += p;
        lds_p[wib][row_lo + r][half * 16 + (l & 15)] = (bf16)p;
      }
    }
    __builtin_amdgcn_wave_barrier();   // order ds_store -> ds_load (same wave)
    const bf16* lp = &lds_p[wib][l & 15][0];
    v8bf plo = *(const v8bf*)(lp + row_lo);
    v8bf phi = *(const v8bf*)(lp + row_lo + 16);
    v16bf pa = make_v16(plo, phi);
#pragma unroll
    for (int dt = 0; dt < 4; ++dt)
      oacc[dt] = wmma_bf16(pa, fragB(vtb, 4096, dt * 16, jc * 32), oacc[dt]);
    __builtin_amdgcn_wave_barrier();   // order ds_load -> next iter's ds_store
  }

  // row sums: reduce the 16 columns within each lane-half
#pragma unroll
  for (int r = 0; r < 8; ++r)
#pragma unroll
    for (int m = 1; m < 16; m <<= 1) rs[r] += __shfl_xor(rs[r], m);

#pragma unroll
  for (int r = 0; r < 8; ++r) {
    int i = it * 16 + row_lo + r;
    float inv_den = 1.0f / (rs[r] + 4096.0f * EPSV);
#pragma unroll
    for (int dt = 0; dt < 4; ++dt) {
      int dd = dt * 16 + (l & 15);
      float num = oacc[dt][r] + EPSV * vsum[bh * 64 + dd];
      AO[((size_t)(b * 1024 + i)) * 512 + h * 64 + dd] = (bf16)(num * inv_den);
    }
  }
}

// ---------------------------------------------------------------------------
// Host-side launch
// ---------------------------------------------------------------------------
extern "C" void kernel_launch(void* const* d_in, const int* in_sizes, int n_in,
                              void* d_out, int out_size, void* d_ws, size_t ws_size,
                              hipStream_t stream) {
  const float* sink    = (const float*)d_in[0];
  const float* source  = (const float*)d_in[1];
  const float* gamma_s = (const float*)d_in[2];
  const float* beta_s  = (const float*)d_in[3];
  const float* gamma_c = (const float*)d_in[4];
  const float* beta_c  = (const float*)d_in[5];
  const float* Wq      = (const float*)d_in[6];
  const float* bq      = (const float*)d_in[7];
  const float* Wkv     = (const float*)d_in[8];
  const float* bkv     = (const float*)d_in[9];
  const float* Wo      = (const float*)d_in[10];
  const float* bo      = (const float*)d_in[11];
  float* out = (float*)d_out;

  char* ws = (char*)d_ws;
  size_t off = 0;
  auto alloc = [&](size_t bytes) -> void* {
    void* p = ws + off;
    off = (off + bytes + 255) & ~(size_t)255;
    return p;
  };
  bf16* WqT    = (bf16*)alloc((size_t)512 * 512 * 2);
  bf16* WkvT   = (bf16*)alloc((size_t)1024 * 512 * 2);
  bf16* WoT    = (bf16*)alloc((size_t)512 * 512 * 2);
  bf16* SLN    = (bf16*)alloc((size_t)2048 * 512 * 2);
  bf16* CLN    = (bf16*)alloc((size_t)8192 * 512 * 2);
  bf16* Qb     = (bf16*)alloc((size_t)2048 * 512 * 2);
  bf16* Kb     = (bf16*)alloc((size_t)8192 * 512 * 2);
  bf16* VT     = (bf16*)alloc((size_t)8192 * 512 * 2);   // [2][8][64][4096]
  bf16* AO     = (bf16*)alloc((size_t)2048 * 512 * 2);
  float* cmax  = (float*)alloc((size_t)65536 * 4);
  float* csum  = (float*)alloc((size_t)65536 * 4);
  float* vsum  = (float*)alloc((size_t)1024 * 4);

  transpose_to_bf16_kernel<<<(512 * 512) / 256, 256, 0, stream>>>(Wq, WqT, 512);
  transpose_to_bf16_kernel<<<(1024 * 512) / 256, 256, 0, stream>>>(Wkv, WkvT, 1024);
  transpose_to_bf16_kernel<<<(512 * 512) / 256, 256, 0, stream>>>(Wo, WoT, 512);

  ln_to_bf16_kernel<<<2048 / 8, 256, 0, stream>>>(sink, gamma_s, beta_s, SLN, 2048);
  ln_to_bf16_kernel<<<8192 / 8, 256, 0, stream>>>(source, gamma_c, beta_c, CLN, 8192);

  gemm_bf16_wmma_kernel<bf16><<<dim3(8, 8), 256, 0, stream>>>(SLN, WqT, bq, Qb, 512, 512);
  kv_gemm_kernel<<<dim3(32, 16), 256, 0, stream>>>(CLN, WkvT, bkv, Kb, VT);

  col_stats_kernel<<<1024, 128, 0, stream>>>(Qb, Kb, cmax, csum);
  vsum_kernel<<<128, 256, 0, stream>>>(VT, vsum);
  attn_out_kernel<<<256, 128, 0, stream>>>(Qb, Kb, VT, cmax, csum, vsum, AO);

  gemm_bf16_wmma_kernel<float><<<dim3(8, 8), 256, 0, stream>>>(AO, WoT, bo, out, 512, 512);
}
